// HierarchicalZ2_12Quantizer_44384192037444
// MI455X (gfx1250) — compile-verified, matrix operand order checked by reading, not run on hardware
//
#include <hip/hip_runtime.h>
#include <hip/hip_bf16.h>
#include <math.h>

typedef __attribute__((ext_vector_type(16))) __bf16 v16bf;
typedef __attribute__((ext_vector_type(8)))  float  v8f;

#define D_MODEL 1024
#define N_TOK   32768          // 8 * 4096
#define TOK_PER_BLOCK 128
#define N_BLOCKS (N_TOK / TOK_PER_BLOCK)   // 256
// packed B: 2 ntiles x 32 ksteps x 32 lanes x 16 bf16 = 32768 bf16 = 64 KB
#define BPACK_ELEMS (2 * 32 * 32 * 16)

// ---------------------------------------------------------------------------
// Kernel 0: pack [w_gate | w_to_l1 | w_to_l2 | w_to_l3u | w_to_l3l] (1024x24,
// padded to 1024x32) into bf16 in the wave32 WMMA B-operand layout.
// Per ISA 16-bit operand layout: lane holds col n = lane&15; lanes 0-15 carry
// K = base+0..7 (elems 0..7) and base+16..23 (elems 8..15); lanes 16-31 carry
// K = base+8..15 and base+24..31.
// ---------------------------------------------------------------------------
__global__ __launch_bounds__(256) void pack_b_kernel(
    const float* __restrict__ w_gate, const float* __restrict__ w_to_l1,
    const float* __restrict__ w_to_l2, const float* __restrict__ w_to_l3u,
    const float* __restrict__ w_to_l3l, __bf16* __restrict__ bpack) {
  int idx = blockIdx.x * 256 + threadIdx.x;      // 0 .. 32767
  int j    = idx & 15;
  int lane = (idx >> 4) & 31;
  int kk   = (idx >> 9) & 31;
  int nt   = idx >> 14;                          // 0 or 1
  int n     = nt * 16 + (lane & 15);
  int khalf = lane >> 4;
  int klocal = (j < 8) ? (khalf * 8 + j) : (16 + khalf * 8 + (j - 8));
  int k = kk * 32 + klocal;                      // 0 .. 1023
  float v = 0.0f;
  if      (n < 3)  v = w_gate  [k * 3 + n];
  else if (n < 6)  v = w_to_l1 [k * 3 + (n - 3)];
  else if (n < 12) v = w_to_l2 [k * 6 + (n - 6)];
  else if (n < 18) v = w_to_l3u[k * 6 + (n - 12)];
  else if (n < 24) v = w_to_l3l[k * 6 + (n - 18)];
  bpack[idx] = (__bf16)v;
}

// ---------------------------------------------------------------------------
// Soft quantize: q = softmax(-cdist(z, codes)/temp) @ codes
// ---------------------------------------------------------------------------
template <int K, int N>
__device__ inline void soft_quant(const float* z, const float* __restrict__ codes,
                                  float invtemp, float* q) {
  float logit[K];
  float mx = -3.0e38f;
#pragma unroll
  for (int c = 0; c < K; ++c) {
    float d2 = 0.0f;
#pragma unroll
    for (int i = 0; i < N; ++i) { float d = z[i] - codes[c * N + i]; d2 += d * d; }
    float l = -sqrtf(d2) * invtemp;
    logit[c] = l;
    mx = fmaxf(mx, l);
  }
  float s = 0.0f;
#pragma unroll
  for (int c = 0; c < K; ++c) { float e = expf(logit[c] - mx); logit[c] = e; s += e; }
  float inv = 1.0f / s;
#pragma unroll
  for (int i = 0; i < N; ++i) q[i] = 0.0f;
#pragma unroll
  for (int c = 0; c < K; ++c) {
    float wc = logit[c] * inv;
#pragma unroll
    for (int i = 0; i < N; ++i) q[i] += wc * codes[c * N + i];
  }
}

__device__ inline float clamp_temp(float log_t) {
  return fminf(fmaxf(expf(log_t), 0.01f), 5.0f);
}

// ---------------------------------------------------------------------------
// Fused kernel: WMMA down-projection -> soft quantize -> fused epilogue.
// 256 threads = 8 waves; each wave owns 16 tokens and both 16-col N tiles.
// ---------------------------------------------------------------------------
__global__ __launch_bounds__(256) void fused_kernel(
    const float* __restrict__ x,
    const float* __restrict__ codes_l1,   // (8,3)
    const float* __restrict__ codes_l2,   // (64,6)
    const float* __restrict__ wf1,        // (3,1024)
    const float* __restrict__ wf2,        // (6,1024)
    const float* __restrict__ wf3,        // (12,1024)
    const float* __restrict__ b_gate,     // (3,)
    const float* __restrict__ temp_l1, const float* __restrict__ temp_l2,
    const float* __restrict__ temp_l3,
    const float* __restrict__ scale_l1, const float* __restrict__ scale_l2,
    const float* __restrict__ scale_l3,
    const __bf16* __restrict__ bpack,
    float* __restrict__ y,
    float* __restrict__ gate_partials) {
  __shared__ float zlds[TOK_PER_BLOCK * 32];   // 16 KB
  __shared__ float qc[TOK_PER_BLOCK * 21];     // ~10.5 KB: gate*scale folded coeffs
  __shared__ float gbuf[TOK_PER_BLOCK * 3];    // gate values for reduction

  const int tid  = threadIdx.x;
  const int wave = tid >> 5;
  const int lane = tid & 31;
  const int blockTok = blockIdx.x * TOK_PER_BLOCK;

  // ------------------ Phase 1: Z = X @ Bpack via WMMA bf16 ------------------
  {
    const int rowTok = blockTok + wave * 16 + (lane & 15);
    const int khalf  = lane >> 4;
    const float* __restrict__ xrow = x + (size_t)rowTok * D_MODEL;
    const v16bf* __restrict__ bp = (const v16bf*)bpack;

    v8f acc0 = {};
    v8f acc1 = {};
#pragma unroll 4
    for (int kk = 0; kk < 32; ++kk) {
      const int kb = kk * 32 + khalf * 8;
      float4 f0 = *(const float4*)(xrow + kb);
      float4 f1 = *(const float4*)(xrow + kb + 4);
      float4 f2 = *(const float4*)(xrow + kb + 16);
      float4 f3 = *(const float4*)(xrow + kb + 20);
      v16bf a;
      a[0]  = (__bf16)f0.x; a[1]  = (__bf16)f0.y; a[2]  = (__bf16)f0.z; a[3]  = (__bf16)f0.w;
      a[4]  = (__bf16)f1.x; a[5]  = (__bf16)f1.y; a[6]  = (__bf16)f1.z; a[7]  = (__bf16)f1.w;
      a[8]  = (__bf16)f2.x; a[9]  = (__bf16)f2.y; a[10] = (__bf16)f2.z; a[11] = (__bf16)f2.w;
      a[12] = (__bf16)f3.x; a[13] = (__bf16)f3.y; a[14] = (__bf16)f3.z; a[15] = (__bf16)f3.w;
      v16bf b0 = bp[(0 * 32 + kk) * 32 + lane];
      v16bf b1 = bp[(1 * 32 + kk) * 32 + lane];
      acc0 = __builtin_amdgcn_wmma_f32_16x16x32_bf16(false, a, false, b0,
                                                     (short)0, acc0, false, false);
      acc1 = __builtin_amdgcn_wmma_f32_16x16x32_bf16(false, a, false, b1,
                                                     (short)0, acc1, false, false);
    }
    // C layout: lane l holds col n=l&15, VGPR r holds row m = r + 8*(l>>4)
    const int n0 = lane & 15;
    const int mbase = wave * 16 + (lane >> 4) * 8;
#pragma unroll
    for (int r = 0; r < 8; ++r) {
      zlds[(mbase + r) * 32 + n0]      = acc0[r];
      zlds[(mbase + r) * 32 + 16 + n0] = acc1[r];
    }
  }
  __syncthreads();

  // ------------------ Phase 2: per-token gate + soft quantize ---------------
  if (tid < TOK_PER_BLOCK) {
    const float* zr = &zlds[tid * 32];
    // gate = softmax(z[0:3] + b_gate)
    float g0 = zr[0] + b_gate[0];
    float g1 = zr[1] + b_gate[1];
    float g2 = zr[2] + b_gate[2];
    float gm = fmaxf(g0, fmaxf(g1, g2));
    float e0 = expf(g0 - gm), e1 = expf(g1 - gm), e2 = expf(g2 - gm);
    float gs = 1.0f / (e0 + e1 + e2);
    g0 = e0 * gs; g1 = e1 * gs; g2 = e2 * gs;

    const float it1 = 1.0f / clamp_temp(temp_l1[0]);
    const float it2 = 1.0f / clamp_temp(temp_l2[0]);
    const float it3 = 1.0f / clamp_temp(temp_l3[0]);

    float q1[3], q2[6], q3u[6], q3l[6];
    soft_quant<8, 3>(zr + 3, codes_l1, it1, q1);
    soft_quant<64, 6>(zr + 6, codes_l2, it2, q2);
    soft_quant<64, 6>(zr + 12, codes_l2, it3, q3u);
    soft_quant<64, 6>(zr + 18, codes_l2, it3, q3l);

    const float c1 = g0 * scale_l1[0];
    const float c2 = g1 * scale_l2[0];
    const float c3 = g2 * scale_l3[0];
    float* qt = &qc[tid * 21];
#pragma unroll
    for (int i = 0; i < 3; ++i) qt[i] = c1 * q1[i];
#pragma unroll
    for (int i = 0; i < 6; ++i) qt[3 + i] = c2 * q2[i];
#pragma unroll
    for (int i = 0; i < 6; ++i) qt[9 + i] = c3 * q3u[i];
#pragma unroll
    for (int i = 0; i < 6; ++i) qt[15 + i] = c3 * q3l[i];

    gbuf[tid * 3 + 0] = g0;
    gbuf[tid * 3 + 1] = g1;
    gbuf[tid * 3 + 2] = g2;
  }
  __syncthreads();

  // deterministic per-block gate partial sums
  if (tid < 3) {
    float s = 0.0f;
    for (int t = 0; t < TOK_PER_BLOCK; ++t) s += gbuf[t * 3 + tid];
    gate_partials[blockIdx.x * 3 + tid] = s;
  }

  // ------------------ Phase 3: epilogue y = x + sum(coef * w_from) ----------
#pragma unroll
  for (int rep = 0; rep < 4; ++rep) {
    const int d = tid + rep * 256;
    float w1v[3], w2v[6], w3v[12];
#pragma unroll
    for (int i = 0; i < 3; ++i) w1v[i] = wf1[i * D_MODEL + d];
#pragma unroll
    for (int i = 0; i < 6; ++i) w2v[i] = wf2[i * D_MODEL + d];
#pragma unroll
    for (int i = 0; i < 12; ++i) w3v[i] = wf3[i * D_MODEL + d];
    for (int t = 0; t < TOK_PER_BLOCK; ++t) {
      const float* qt = &qc[t * 21];
      float o = 0.0f;
#pragma unroll
      for (int i = 0; i < 3; ++i) o += qt[i] * w1v[i];
#pragma unroll
      for (int i = 0; i < 6; ++i) o += qt[3 + i] * w2v[i];
#pragma unroll
      for (int i = 0; i < 12; ++i) o += qt[9 + i] * w3v[i];
      const size_t base = (size_t)(blockTok + t) * D_MODEL + d;
      y[base] = x[base] + o;
    }
  }
}

// ---------------------------------------------------------------------------
// Kernel 2: reduce gate partials -> 3 gate means at tail of d_out
// ---------------------------------------------------------------------------
__global__ void reduce_gate_kernel(const float* __restrict__ gate_partials,
                                   float* __restrict__ out_tail) {
  int i = threadIdx.x;
  if (i < 3) {
    float s = 0.0f;
    for (int b = 0; b < N_BLOCKS; ++b) s += gate_partials[b * 3 + i];
    out_tail[i] = s * (1.0f / (float)N_TOK);
  }
}

extern "C" void kernel_launch(void* const* d_in, const int* in_sizes, int n_in,
                              void* d_out, int out_size, void* d_ws, size_t ws_size,
                              hipStream_t stream) {
  const float* x        = (const float*)d_in[0];
  const float* codes_l1 = (const float*)d_in[1];
  const float* codes_l2 = (const float*)d_in[2];
  const float* w_to_l1  = (const float*)d_in[3];
  const float* w_from_l1= (const float*)d_in[4];
  const float* w_to_l2  = (const float*)d_in[5];
  const float* w_from_l2= (const float*)d_in[6];
  const float* w_to_l3u = (const float*)d_in[7];
  const float* w_to_l3l = (const float*)d_in[8];
  const float* w_from_l3= (const float*)d_in[9];
  const float* w_gate   = (const float*)d_in[10];
  const float* b_gate   = (const float*)d_in[11];
  const float* temp_l1  = (const float*)d_in[12];
  const float* temp_l2  = (const float*)d_in[13];
  const float* temp_l3  = (const float*)d_in[14];
  const float* scale_l1 = (const float*)d_in[15];
  const float* scale_l2 = (const float*)d_in[16];
  const float* scale_l3 = (const float*)d_in[17];

  __bf16* bpack = (__bf16*)d_ws;
  float* gate_partials = (float*)((char*)d_ws + BPACK_ELEMS * sizeof(__bf16));

  float* y = (float*)d_out;
  float* out_tail = y + (size_t)N_TOK * D_MODEL;

  pack_b_kernel<<<BPACK_ELEMS / 256, 256, 0, stream>>>(
      w_gate, w_to_l1, w_to_l2, w_to_l3u, w_to_l3l, bpack);

  fused_kernel<<<N_BLOCKS, 256, 0, stream>>>(
      x, codes_l1, codes_l2, w_from_l1, w_from_l2, w_from_l3, b_gate,
      temp_l1, temp_l2, temp_l3, scale_l1, scale_l2, scale_l3,
      bpack, y, gate_partials);

  reduce_gate_kernel<<<1, 32, 0, stream>>>(gate_partials, out_tail);
}
